// GraphAttentionConv_73169062855371
// MI455X (gfx1250) — matrix operand order repaired and checked by reading, not compiled, not run on hardware
//
#include <hip/hip_runtime.h>

// GAT layer: N=8192 nodes, IN_F=512, OUT_F=256
// out = softmax_row(mask(leaky_relu(e1_i + e2_j), adj)) @ (X@W) + bias
//
// Single-pass softmax trick: lrelu is monotonic, so
//   mhat_i = lrelu(e1_i + max_j e2_j)  >=  every masked score in row i.
// Using this fixed upper bound (instead of the true masked row max) keeps
// exp(s - mhat) in (0,1] and cancels exactly in the softmax ratio, so the
// aggregation kernel streams adj ONCE (256 MB @ 23.3 TB/s ~ 11 us) while
// doing the 34 GFLOP attention GEMM on v_wmma_f32_16x16x32_f16.

#define N_NODES 8192
#define IN_FEAT 512
#define OUT_FEAT 256
#define NEG_INF_F (-9e15f)
#define SLOPE 0.25f

typedef _Float16 half8  __attribute__((ext_vector_type(8)));
typedef _Float16 half16 __attribute__((ext_vector_type(16)));
typedef float    float8 __attribute__((ext_vector_type(8)));

#define SHUF16(lo, hi) __builtin_shufflevector(lo, hi, \
    0,1,2,3,4,5,6,7,8,9,10,11,12,13,14,15)

// ---------------------------------------------------------------------------
// K1a: f32 -> f16 convert (grid-stride)
__global__ void gat_cvt_f16(const float* __restrict__ src,
                            _Float16* __restrict__ dst, int n) {
    int i = blockIdx.x * blockDim.x + threadIdx.x;
    int stride = gridDim.x * blockDim.x;
    for (; i < n; i += stride) dst[i] = (_Float16)src[i];
}

// K1b: W[k][n] (f32) -> wT[n][k] (f16), so WMMA B-fragments are contiguous
__global__ void gat_cvt_wT(const float* __restrict__ w,
                           _Float16* __restrict__ wT) {
    int idx = blockIdx.x * blockDim.x + threadIdx.x;
    if (idx < IN_FEAT * OUT_FEAT) {
        int k = idx / OUT_FEAT, n = idx % OUT_FEAT;
        wT[n * IN_FEAT + k] = (_Float16)w[idx];
    }
}

// ---------------------------------------------------------------------------
// K2: h = X @ W via v_wmma_f32_16x16x32_f16.  Stores hT[col][row] in f16.
// Block = 128 thr (4 waves).  Block tile: 16 rows x 64 cols (1 col-tile/wave).
__global__ void __launch_bounds__(128)
gat_gemm_h(const _Float16* __restrict__ xF16,   // [N, IN_FEAT]
           const _Float16* __restrict__ wT,     // [OUT_FEAT, IN_FEAT]
           _Float16* __restrict__ hT) {         // [OUT_FEAT, N]
    const int tid  = threadIdx.x;
    const int wave = tid >> 5;
    const int lane = tid & 31;
    const int hi16 = lane >> 4;     // 0|1: lane half-group
    const int l15  = lane & 15;
    const int rowBase = blockIdx.x * 16;
    const int colBase = blockIdx.y * 64 + wave * 16;

    // A-frag (16x32 f16): lane<16 -> row=lane, K {0..15}
    //                     lane>=16 -> row=lane-16, K {16..31}
    const _Float16* aRow = xF16 + (size_t)(rowBase + l15) * IN_FEAT + (hi16 << 4);
    // B-frag (32x16 f16): 16 contiguous K starting at (hi16*16), col = l15
    const _Float16* bRow = wT + (size_t)(colBase + l15) * IN_FEAT + (hi16 << 4);

    float8 acc = {};
    for (int kb = 0; kb < IN_FEAT; kb += 32) {
        half8 a_lo = *(const half8*)(aRow + kb);
        half8 a_hi = *(const half8*)(aRow + kb + 8);
        half16 A = SHUF16(a_lo, a_hi);
        half16 B = *(const half16*)(bRow + kb);
        acc = __builtin_amdgcn_wmma_f32_16x16x32_f16(
            false, A, false, B, (short)0, acc, false, false);
    }
    // C layout: VGPR v -> M = v + 8*hi16, N = l15
    for (int v = 0; v < 8; ++v) {
        int row = rowBase + v + (hi16 << 3);
        int col = colBase + l15;
        hT[(size_t)col * N_NODES + row] = (_Float16)acc[v];
    }
}

// ---------------------------------------------------------------------------
// K3: e1[i] = sum_c h[i][c]*a1[c], e2 likewise.  One thread per node; reads
// of hT[c][i] are coalesced across threads for each c.
__global__ void __launch_bounds__(256)
gat_evec(const _Float16* __restrict__ hT, const float* __restrict__ a1,
         const float* __restrict__ a2, float* __restrict__ e1,
         float* __restrict__ e2) {
    int i = blockIdx.x * blockDim.x + threadIdx.x;
    float s1 = 0.f, s2 = 0.f;
    for (int c = 0; c < OUT_FEAT; ++c) {
        float hv = (float)hT[(size_t)c * N_NODES + i];
        s1 += hv * a1[c];
        s2 += hv * a2[c];
    }
    e1[i] = s1;
    e2[i] = s2;
}

// ---------------------------------------------------------------------------
// K4: global max of e2 (8192 values -> 1 scalar), single block.
__global__ void __launch_bounds__(256)
gat_e2max(const float* __restrict__ e2, float* __restrict__ outmax) {
    const int tid = threadIdx.x;
    float m = -3.4e38f;
    for (int q = 0; q < N_NODES / 256; ++q)
        m = fmaxf(m, e2[q * 256 + tid]);
    for (int off = 16; off; off >>= 1) m = fmaxf(m, __shfl_xor(m, off, 32));
    __shared__ float red[8];
    if ((tid & 31) == 0) red[tid >> 5] = m;
    __syncthreads();
    if (tid == 0) {
        float r = red[0];
        for (int w = 1; w < 8; ++w) r = fmaxf(r, red[w]);
        outmax[0] = r;
    }
}

// ---------------------------------------------------------------------------
// K5 (fused, single adj pass):
//   p_ij = exp(lrelu(e1_i+e2_j) - mhat_i) if adj_ij>0 else 0
//   out[i,:] = (sum_j p_ij h[j,:]) / (sum_j p_ij) + bias
// Block = 512 thr (16 waves): 16-row i-block x all 256 cols.
// Wave w computes probability row i=w into LDS (64 j per iter, f16), owns
// output col strip [16w,16w+16), does 2 WMMAs/iter, and accumulates its
// row's normalizer from the same f16-rounded p the WMMAs consume.
__global__ void __launch_bounds__(512)
gat_aggregate(const int* __restrict__ adj, const _Float16* __restrict__ hT,
              const float* __restrict__ e1, const float* __restrict__ e2,
              const float* __restrict__ e2maxp, const float* __restrict__ bias,
              float* __restrict__ out) {
    alignas(32) __shared__ _Float16 sP[2][16][64];   // double-buffered p tile
    __shared__ float sL[16];                         // per-row normalizers

    const int tid  = threadIdx.x;
    const int wave = tid >> 5;
    const int lane = tid & 31;
    const int hi16 = lane >> 4;
    const int l15  = lane & 15;
    const int iBase   = blockIdx.x * 16;
    const int colBase = wave * 16;
    const int myi     = iBase + wave;

    const float e1i = e1[myi];
    const float t0  = e1i + e2maxp[0];
    const float mhat = t0 > 0.f ? t0 : SLOPE * t0;   // upper bound on row scores
    const int* __restrict__ arow = adj + (size_t)myi * N_NODES;
    const _Float16* bRow = hT + (size_t)(colBase + l15) * N_NODES + (hi16 << 4);

    float8 acc = {};
    float lsum = 0.f;
    for (int jc = 0; jc < N_NODES / 64; ++jc) {
        const int jBase = jc * 64;
        // --- probabilities for (row=wave, cols jBase+lane, jBase+32+lane)
        int   a0 = arow[jBase + lane];
        int   a1 = arow[jBase + 32 + lane];
        float u0 = e1i + e2[jBase + lane];
        float u1 = e1i + e2[jBase + 32 + lane];
        u0 = u0 > 0.f ? u0 : SLOPE * u0;
        u1 = u1 > 0.f ? u1 : SLOPE * u1;
        float p0 = (a0 > 0) ? __expf(u0 - mhat) : 0.0f;
        float p1 = (a1 > 0) ? __expf(u1 - mhat) : 0.0f;
        _Float16 ph0 = (_Float16)p0;
        _Float16 ph1 = (_Float16)p1;
        lsum += (float)ph0 + (float)ph1;             // normalizer == what WMMA sees
        const int buf = jc & 1;
        sP[buf][wave][lane]      = ph0;
        sP[buf][wave][lane + 32] = ph1;
        if (jc + 1 < N_NODES / 64)                   // cover next 256B chunk
            __builtin_prefetch(arow + jBase + 64 + 2 * lane, 0, 1);
        __syncthreads();

        // --- A-frag gathers from LDS (contiguous k = 16*hi16 + s, same order as B)
        half8 a_lo0 = *(const half8*)&sP[buf][l15][(hi16 << 4)];
        half8 a_hi0 = *(const half8*)&sP[buf][l15][(hi16 << 4) + 8];
        half8 a_lo1 = *(const half8*)&sP[buf][l15][32 + (hi16 << 4)];
        half8 a_hi1 = *(const half8*)&sP[buf][l15][32 + (hi16 << 4) + 8];
        half16 A0 = SHUF16(a_lo0, a_hi0);
        half16 A1 = SHUF16(a_lo1, a_hi1);
        // --- B-frags: 16 contiguous node-indices from hT (32B-aligned)
        half16 B0 = *(const half16*)(bRow + jBase);
        half16 B1 = *(const half16*)(bRow + jBase + 32);

        acc = __builtin_amdgcn_wmma_f32_16x16x32_f16(
            false, A0, false, B0, (short)0, acc, false, false);
        acc = __builtin_amdgcn_wmma_f32_16x16x32_f16(
            false, A1, false, B1, (short)0, acc, false, false);
    }
    // publish per-row normalizer: wave w owns row w
    for (int off = 16; off; off >>= 1) lsum += __shfl_xor(lsum, off, 32);
    if (lane == 0) sL[wave] = lsum;
    __syncthreads();
    for (int v = 0; v < 8; ++v) {
        int r = v + (hi16 << 3);                     // row within i-block
        float invL = 1.0f / fmaxf(sL[r], 1e-30f);
        out[(size_t)(iBase + r) * OUT_FEAT + colBase + l15] =
            acc[v] * invL + bias[colBase + l15];
    }
}

// ---------------------------------------------------------------------------
extern "C" void kernel_launch(void* const* d_in, const int* in_sizes, int n_in,
                              void* d_out, int out_size, void* d_ws, size_t ws_size,
                              hipStream_t stream) {
    const float* d_input = (const float*)d_in[0];   // [8192, 512] f32
    const int*   d_adj   = (const int*)d_in[1];     // [8192, 8192] i32
    const float* d_w     = (const float*)d_in[2];   // [512, 256] f32
    const float* d_a1    = (const float*)d_in[3];   // [256, 1] f32
    const float* d_a2    = (const float*)d_in[4];   // [256, 1] f32
    const float* d_bias  = (const float*)d_in[5];   // [256] f32
    float* out = (float*)d_out;                     // [8192, 256] f32

    // workspace carve-up (~12.3 MB)
    char* ws = (char*)d_ws;
    _Float16* xF16 = (_Float16*)(ws);                                   // 8 MB
    _Float16* wT   = (_Float16*)(ws + (size_t)8 * 1024 * 1024);         // 256 KB
    _Float16* hT   = (_Float16*)(ws + (size_t)8 * 1024 * 1024 + 256 * 1024); // 4 MB
    float* e1  = (float*)(ws + (size_t)12 * 1024 * 1024 + 256 * 1024);
    float* e2  = e1 + N_NODES;
    float* e2m = e2 + N_NODES;

    gat_cvt_f16<<<1024, 256, 0, stream>>>(d_input, xF16, N_NODES * IN_FEAT);
    gat_cvt_wT<<<(IN_FEAT * OUT_FEAT + 255) / 256, 256, 0, stream>>>(d_w, wT);
    gat_gemm_h<<<dim3(N_NODES / 16, OUT_FEAT / 64), 128, 0, stream>>>(xF16, wT, hT);
    gat_evec<<<N_NODES / 256, 256, 0, stream>>>(hT, d_a1, d_a2, e1, e2);
    gat_e2max<<<1, 256, 0, stream>>>(e2, e2m);
    gat_aggregate<<<N_NODES / 16, 512, 0, stream>>>(d_adj, hT, e1, e2, e2m,
                                                    d_bias, out);
}
